// ResBlock_12979391169046
// MI455X (gfx1250) — matrix operand order repaired and checked by reading, not compiled, not run on hardware
//
#include <hip/hip_runtime.h>

#define NVOX   200000
#define KOFF   27
#define CH     32
#define NTILES (NVOX / 16)
#define EPSF   1e-5f
#define NBLK   512        // conv grid: 8 waves/block, grid-stride over tiles

typedef __attribute__((ext_vector_type(16))) _Float16 v16h;
typedef __attribute__((ext_vector_type(8)))  _Float16 v8h;
typedef __attribute__((ext_vector_type(8)))  float    v8f;

union BV { v16h v; v8h h[2]; };

// ---------------------------------------------------------------------------
// prep: features f32->f16 (+ one zeroed row at index NVOX used to realize the
// neighbor mask as a gather redirect); W1/W2 -> f16, transposed to
// [k][out d][K-slot s] with K-slot s = input channel (no permutation: B uses
// the same contiguous k order as A) so the WMMA A-fragment is a single contiguous 32B load per
// lane. Zero the h16 zero-row and the BN accumulators.
// ---------------------------------------------------------------------------
__global__ __launch_bounds__(256) void prep_kernel(
    const float* __restrict__ feat, _Float16* __restrict__ feat16,
    _Float16* __restrict__ h16,
    const float* __restrict__ W1,   const float* __restrict__ W2,
    _Float16* __restrict__ W1t,     _Float16* __restrict__ W2t,
    float* __restrict__ sums1,      float* __restrict__ sums2)
{
    int i = blockIdx.x * blockDim.x + threadIdx.x;
    if (i < (NVOX + 1) * CH)
        feat16[i] = (i < NVOX * CH) ? (_Float16)feat[i] : (_Float16)0.f;
    if (i < CH) h16[NVOX * CH + i] = (_Float16)0.f;   // zero row for conv2
    if (i < 2 * KOFF * CH * CH) {
        int which = i / (KOFF * CH * CH);
        int u = i - which * (KOFF * CH * CH);
        int k = u >> 10;            // offset
        int r = u & 1023;
        int c = r >> 5;             // input channel
        int d = r & 31;             // output channel
        int grp = c >> 3;
        int s = (grp == 1 || grp == 2) ? c : c;   // K-slot = input channel (no permutation)
        _Float16 v = (_Float16)(which ? W2[u] : W1[u]);
        (which ? W2t : W1t)[k * 1024 + d * 32 + s] = v;
    }
    if (i < 64) { sums1[i] = 0.f; sums2[i] = 0.f; }
}

// ---------------------------------------------------------------------------
// Submanifold conv via v_wmma_f32_16x16x32_f16.
// One wave32 = one 16-voxel tile (grid-stride); W staged in LDS with XOR
// chunk swizzle for conflict-free ds_load_b128. Mask folded into the index
// (masked-off neighbors gather the zeroed row NVOX) so the inner loop is
// pure load+WMMA. Per-channel BN sums reduced through LDS, one global
// atomicAdd per channel per block.
// ---------------------------------------------------------------------------
__global__ __launch_bounds__(256) void subm_conv_wmma(
    const _Float16* __restrict__ feat16,        // [N+1][32] f16 (row N = zeros)
    const int* __restrict__ nbr_idx,            // [N][27]
    const unsigned char* __restrict__ nbr_mask, // [N][27] (bool, 1 byte)
    const _Float16* __restrict__ Wt,            // [27][32 d][32 s] f16
    float* __restrict__ y,                      // [N][32] f32
    float* __restrict__ sums)                   // [64]: 0..31 sum, 32..63 sumsq
{
    __shared__ _Float16 sW[KOFF * CH * CH];     // 55,296 B (swizzled chunks)
    __shared__ float sSum[64];

    int tid = threadIdx.x;
    if (tid < 64) sSum[tid] = 0.f;

    // stage W: rows of 32 halfs, 4 chunks of 8 halfs; swizzle chunk by d
    for (int t = tid; t < KOFF * CH * 4; t += 256) {
        int row = t >> 2, ch = t & 3;
        int d = row & 31;
        int chs = ch ^ ((d >> 2) & 3);
        *(v8h*)(sW + row * 32 + chs * 8) = *((const v8h*)(Wt + row * 32) + ch);
    }
    __syncthreads();

    int lane = tid & 31;
    int l16  = lane & 15;       // A: voxel row  /  B,D: column
    int hsel = lane >> 4;       // lane-half select per ISA fragment layouts
    int sw   = (l16 >> 2) & 3;  // LDS chunk swizzle
    int c0   = (hsel * 2) ^ sw; // swizzled chunk for fragment halfs 0..7
    int c1   = c0 ^ 1;          // swizzled chunk for fragment halfs 8..15

    float s0 = 0.f, q0 = 0.f, s1 = 0.f, q1 = 0.f;

    for (int tile = blockIdx.x * 8 + (tid >> 5); tile < NTILES;
         tile += (int)gridDim.x * 8) {
        int vox = tile * 16 + l16;

        // hoist neighbor indices; fold mask into index (zero-row redirect)
        const int* ip = nbr_idx + (size_t)vox * KOFF;
        const unsigned char* mp = nbr_mask + (size_t)vox * KOFF;
        int nidx[KOFF];
#pragma unroll
        for (int k = 0; k < KOFF; ++k)
            nidx[k] = mp[k] ? ip[k] : NVOX;

        v8f acc0 = {};   // output channels 0..15
        v8f acc1 = {};   // output channels 16..31

#pragma unroll
        for (int k = 0; k < KOFF; ++k) {
            // A fragment: lane-half h holds channels h*16..h*16+15 (one 32B load)
            v16h a = *(const v16h*)(feat16 + (size_t)nidx[k] * CH + hsel * 16);

            // B fragments from LDS (swizzled chunks)
            const _Float16* wb0 = sW + ((size_t)k * CH + l16) * CH;
            const _Float16* wb1 = sW + ((size_t)k * CH + l16 + 16) * CH;
            BV b0, b1;
            b0.h[0] = *(const v8h*)(wb0 + c0 * 8);
            b0.h[1] = *(const v8h*)(wb0 + c1 * 8);
            b1.h[0] = *(const v8h*)(wb1 + c0 * 8);
            b1.h[1] = *(const v8h*)(wb1 + c1 * 8);

            acc0 = __builtin_amdgcn_wmma_f32_16x16x32_f16(
                false, a, false, b0.v, (short)0, acc0, false, false);
            acc1 = __builtin_amdgcn_wmma_f32_16x16x32_f16(
                false, a, false, b1.v, (short)0, acc1, false, false);
        }

        // D layout: VGPR i holds voxel row (i + 8*hsel), column l16 (+16)
        float* yp = y + (size_t)tile * 16 * CH;
#pragma unroll
        for (int i = 0; i < 8; ++i) {
            int m = i + hsel * 8;
            float v0 = acc0[i], v1 = acc1[i];
            yp[(size_t)m * CH + l16]      = v0;
            yp[(size_t)m * CH + l16 + 16] = v1;
            s0 += v0; q0 += v0 * v0;
            s1 += v1; q1 += v1 * v1;
        }
    }

    atomicAdd(&sSum[l16],      s0);
    atomicAdd(&sSum[16 + l16], s1);
    atomicAdd(&sSum[32 + l16], q0);
    atomicAdd(&sSum[48 + l16], q1);
    __syncthreads();
    if (tid < 64) atomicAdd(&sums[tid], sSum[tid]);
}

// ---------------------------------------------------------------------------
// BN finalize: A = gamma * rsqrt(var+eps), B = beta - mu*A
// ---------------------------------------------------------------------------
__global__ void bn_finalize(const float* __restrict__ sums,
                            const float* __restrict__ gamma,
                            const float* __restrict__ beta,
                            float* __restrict__ A, float* __restrict__ B)
{
    int c = threadIdx.x;
    if (c < CH) {
        float inv_n = 1.0f / (float)NVOX;
        float mu  = sums[c] * inv_n;
        float var = sums[32 + c] * inv_n - mu * mu;
        float a   = rsqrtf(var + EPSF) * gamma[c];
        A[c] = a;
        B[c] = beta[c] - mu * a;
    }
}

// BN + ReLU, emit f16 activations for conv2
__global__ __launch_bounds__(256) void bn_relu_f16(
    const float* __restrict__ y, const float* __restrict__ A,
    const float* __restrict__ B, _Float16* __restrict__ h16)
{
    int i = blockIdx.x * blockDim.x + threadIdx.x;
    if (i < NVOX * CH) {
        int c = i & 31;
        float v = y[i] * A[c] + B[c];
        h16[i] = (_Float16)(v > 0.f ? v : 0.f);
    }
}

// BN + residual + ReLU -> final f32 output
__global__ __launch_bounds__(256) void bn_res_relu(
    const float* __restrict__ y, const float* __restrict__ A,
    const float* __restrict__ B, const float* __restrict__ feat,
    float* __restrict__ out)
{
    int i = blockIdx.x * blockDim.x + threadIdx.x;
    if (i < NVOX * CH) {
        int c = i & 31;
        float v = y[i] * A[c] + B[c] + feat[i];
        out[i] = v > 0.f ? v : 0.f;
    }
}

// ---------------------------------------------------------------------------
extern "C" void kernel_launch(void* const* d_in, const int* in_sizes, int n_in,
                              void* d_out, int out_size, void* d_ws, size_t ws_size,
                              hipStream_t stream)
{
    const float*         features = (const float*)d_in[0];
    const int*           nbr_idx  = (const int*)d_in[1];
    const unsigned char* nbr_mask = (const unsigned char*)d_in[2]; // bool array
    const float*         W1 = (const float*)d_in[3];
    const float*         g1 = (const float*)d_in[4];
    const float*         b1 = (const float*)d_in[5];
    const float*         W2 = (const float*)d_in[6];
    const float*         g2 = (const float*)d_in[7];
    const float*         b2 = (const float*)d_in[8];
    float* out = (float*)d_out;

    // workspace layout (all offsets multiples of 256 B; total ~49 MB)
    char* ws = (char*)d_ws;
    _Float16* feat16 = (_Float16*)(ws);                 // (N+1)*32*2 = 12,800,064 B
    _Float16* h16    = (_Float16*)(ws + 12800256);      // 12,800,064 B
    _Float16* W1t    = (_Float16*)(ws + 25600512);      //     55,296 B
    _Float16* W2t    = (_Float16*)(ws + 25655808);      //     55,296 B
    float*    sums1  = (float*)   (ws + 25711104);      //        256 B
    float*    sums2  = (float*)   (ws + 25711360);      //        256 B
    float*    bnA1   = (float*)   (ws + 25711616);
    float*    bnB1   = (float*)   (ws + 25711872);
    float*    bnA2   = (float*)   (ws + 25712128);
    float*    bnB2   = (float*)   (ws + 25712384);
    float*    ybuf   = (float*)   (ws + 25712640);      // 25,600,000 B

    const int elemGrid = (NVOX * CH + 255) / 256;       // 25000

    prep_kernel<<<elemGrid, 256, 0, stream>>>(features, feat16, h16, W1, W2,
                                              W1t, W2t, sums1, sums2);

    subm_conv_wmma<<<NBLK, 256, 0, stream>>>(feat16, nbr_idx, nbr_mask,
                                             W1t, ybuf, sums1);
    bn_finalize<<<1, 32, 0, stream>>>(sums1, g1, b1, bnA1, bnB1);
    bn_relu_f16<<<elemGrid, 256, 0, stream>>>(ybuf, bnA1, bnB1, h16);

    subm_conv_wmma<<<NBLK, 256, 0, stream>>>(h16, nbr_idx, nbr_mask,
                                             W2t, ybuf, sums2);
    bn_finalize<<<1, 32, 0, stream>>>(sums2, g2, b2, bnA2, bnB2);
    bn_res_relu<<<elemGrid, 256, 0, stream>>>(ybuf, bnA2, bnB2, features, out);
}
